// UP_38637525795023
// MI455X (gfx1250) — matrix operand order repaired and checked by reading, not compiled, not run on hardware
//
#include <hip/hip_runtime.h>

// ---------------------------------------------------------------------------
// Types for CDNA5 WMMA (wave32). v16bf = 16 bf16 = 8 VGPRs (A/B operand),
// v8f = 8 f32 = 8 VGPRs (C/D accumulator).
// ---------------------------------------------------------------------------
typedef __bf16 bf16_t;
typedef __attribute__((ext_vector_type(16))) __bf16 v16bf;
typedef __attribute__((ext_vector_type(8)))  float  v8f;

__device__ __forceinline__ bf16_t f2bf(float f) {
    unsigned u = __builtin_bit_cast(unsigned, f);
    unsigned r = u + 0x7FFFu + ((u >> 16) & 1u);   // round-to-nearest-even
    unsigned short h = (unsigned short)(r >> 16);
    return __builtin_bit_cast(bf16_t, h);
}
__device__ __forceinline__ float bf2f(bf16_t b) {
    unsigned short h = __builtin_bit_cast(unsigned short, b);
    unsigned u = ((unsigned)h) << 16;
    return __builtin_bit_cast(float, u);
}

// Halo-padded NHWC pixel index: valid for y in [-1,H], x in [-1,W].
__device__ __forceinline__ size_t pidx(int n, int y, int x, int H, int W) {
    return (size_t)(n * (H + 2) + (y + 1)) * (W + 2) + (x + 1);
}

// ---------------------------------------------------------------------------
// Weight pre-swizzle: OIHW f32 -> A-operand fragment stream, bf16.
// Layout: wp[((oct*KB + kb)*32 + lane)*16 + e], K order = tap*Cin + c.
// A fragment in this code's contiguous K order: lane<16 row M=lane holds
// K = {kb*32 + e}, lane>=16 same row, K shifted by +16.
// ---------------------------------------------------------------------------
__global__ void k_packw(const float* __restrict__ w, bf16_t* __restrict__ wp,
                        int O, int Cin) {
    int KB  = (Cin * 9) >> 5;
    int OCT = (O + 15) >> 4;
    int total = OCT * KB * 512;
    int t = blockIdx.x * 256 + threadIdx.x;
    if (t >= total) return;
    int e    = t & 15;
    int lane = (t >> 4) & 31;
    int kb   = (t >> 9) % KB;
    int oct  = (t >> 9) / KB;
    int m    = lane & 15;
    int hf   = lane >> 4;
    int kin  = e + (hf ? 16 : 0);
    int Kg   = kb * 32 + kin;
    int oc   = oct * 16 + m;
    float v  = 0.f;
    if (oc < O) {
        int tap = Kg / Cin;
        int c   = Kg - tap * Cin;
        v = w[((size_t)(oc * Cin + c)) * 9 + tap];   // OIHW, tap = ky*3+kx
    }
    wp[t] = f2bf(v);
}

// ---------------------------------------------------------------------------
// Zero the 1-pixel halo border of an NHWC(halo) bf16 buffer.
// ---------------------------------------------------------------------------
__global__ void k_zero_halo(bf16_t* __restrict__ buf, int Cpix, int N, int H, int W) {
    int P = (H + 2) * (W + 2);
    int t = blockIdx.x * 256 + threadIdx.x;
    if (t >= N * P) return;
    int p = t % P;
    int y = p / (W + 2) - 1;
    int x = p % (W + 2) - 1;
    if ((unsigned)y < (unsigned)H && (unsigned)x < (unsigned)W) return;
    bf16_t z = __builtin_bit_cast(bf16_t, (unsigned short)0);
    bf16_t* d = buf + (size_t)t * Cpix;
    for (int c = 0; c < Cpix; ++c) d[c] = z;
}

// ---------------------------------------------------------------------------
// NCHW f32 (xl | xr) -> channels [0..127] of NHWC(halo) bf16 concat buffer.
// ---------------------------------------------------------------------------
__global__ void k_pack3(const float* __restrict__ xl, const float* __restrict__ xr,
                        bf16_t* __restrict__ actb, int N, int H, int W) {
    int HW = H * W;
    int total = N * HW * 128;
    int t = blockIdx.x * 256 + threadIdx.x;
    if (t >= total) return;
    int c   = t & 127;
    int pix = t >> 7;
    int n   = pix / HW;
    int rem = pix - n * HW;
    int y   = rem / W;
    int x   = rem - y * W;
    const float* src = (c < 64) ? xl : xr;
    int cc = c & 63;
    actb[pidx(n, y, x, H, W) * 192 + c] = f2bf(src[((size_t)(n * 64 + cc)) * HW + rem]);
}

// Single NCHW f32 64ch tensor -> channel slice of the NHWC(halo) concat buffer.
__global__ void k_pack1(const float* __restrict__ src, bf16_t* __restrict__ actb,
                        int chOff, int N, int H, int W) {
    int HW = H * W;
    int total = N * HW * 64;
    int t = blockIdx.x * 256 + threadIdx.x;
    if (t >= total) return;
    int c   = t & 63;
    int pix = t >> 6;
    int n   = pix / HW;
    int rem = pix - n * HW;
    int y   = rem / W;
    int x   = rem - y * W;
    actb[pidx(n, y, x, H, W) * 192 + chOff + c] =
        f2bf(src[((size_t)(n * 64 + c)) * HW + rem]);
}

// ---------------------------------------------------------------------------
// Upsample(2x nearest) + 2x2 conv (pad 0) + zero-pad bottom/right to HxW.
// Writes bf16 into channels [128..191] of the NHWC(halo) concat buffer.
// ---------------------------------------------------------------------------
__global__ void k_upconv(const float* __restrict__ xd, const float* __restrict__ w,
                         const float* __restrict__ bias, bf16_t* __restrict__ actb,
                         int N, int H, int W) {
    __shared__ float wsh[128 * 4 * 16];        // [ic][tap][o]
    int ocq  = blockIdx.x & 3;
    int pblk = blockIdx.x >> 2;
    for (int d = threadIdx.x; d < 8192; d += 256) {
        int o   = d & 15;
        int tap = (d >> 4) & 3;
        int ic  = d >> 6;
        int oc  = ocq * 16 + o;
        wsh[d] = w[((size_t)(oc * 128 + ic) * 2 + (tap >> 1)) * 2 + (tap & 1)];
    }
    __syncthreads();
    int HW  = H * W;
    int pix = pblk * 256 + threadIdx.x;
    int n   = pix / HW;
    int rem = pix - n * HW;
    int y   = rem / W;
    int x   = rem - y * W;
    int Hh = H >> 1, Wh = W >> 1;
    float acc[16];
#pragma unroll
    for (int o = 0; o < 16; ++o) acc[o] = bias[ocq * 16 + o];
    if (y < H - 1 && x < W - 1) {
        for (int ic = 0; ic < 128; ++ic) {
            const float* xp = xd + ((size_t)(n * 128 + ic)) * Hh * Wh;
            float v00 = xp[(y >> 1) * Wh + (x >> 1)];
            float v01 = xp[(y >> 1) * Wh + ((x + 1) >> 1)];
            float v10 = xp[((y + 1) >> 1) * Wh + (x >> 1)];
            float v11 = xp[((y + 1) >> 1) * Wh + ((x + 1) >> 1)];
            const float* wr = &wsh[ic * 64];
#pragma unroll
            for (int o = 0; o < 16; ++o)
                acc[o] += v00 * wr[o] + v01 * wr[16 + o] + v10 * wr[32 + o] + v11 * wr[48 + o];
        }
    } else {
#pragma unroll
        for (int o = 0; o < 16; ++o) acc[o] = 0.f;   // pad region
    }
    bf16_t* op = actb + pidx(n, y, x, H, W) * 192 + 128 + ocq * 16;
#pragma unroll
    for (int o = 0; o < 16; ++o) op[o] = f2bf(acc[o]);
}

// ---------------------------------------------------------------------------
// 3x3 conv as implicit GEMM on WMMA bf16, halo-padded input (no predication).
// One wave = 16 output channels x 32 pixels (two 16-px tiles, shared A frags).
// Inner loop: 1x A load (32B/lane) + 2x B load (32B/lane) -> 2x v_wmma.
// ---------------------------------------------------------------------------
__global__ __launch_bounds__(256)
void k_conv3x3(const bf16_t* __restrict__ act, int Cin,
               const bf16_t* __restrict__ wp, const float* __restrict__ bias,
               int OC, int OCT, int relu,
               const float* __restrict__ residual,
               float* __restrict__ outF32, bf16_t* __restrict__ outBf,
               int N, int H, int W) {
    const int tpp     = W >> 5;                  // 32-px pair-tiles per row
    const int npairs  = N * H * tpp;
    int gw   = blockIdx.x * 8 + (threadIdx.x >> 5);
    int lane = threadIdx.x & 31;
    int oct  = gw % OCT;
    int pair = gw / OCT;
    if (pair >= npairs) return;
    int n    = pair / (H * tpp);
    int rem  = pair - n * H * tpp;
    int y    = rem / tpp;
    int x0   = (rem - y * tpp) << 5;
    int col  = lane & 15, khalf = lane >> 4;
    int xA   = x0 + col;
    int xB   = xA + 16;
    int c16h = Cin >> 5;                         // 32-K blocks per tap
    int KB   = (Cin * 9) >> 5;

    v8f accA = {};
    v8f accB = {};
    const bf16_t* wpo = wp + (size_t)oct * KB * 512 + lane * 16;
    int kb = 0;
#pragma unroll
    for (int tap = 0; tap < 9; ++tap) {
        int yy  = y + tap / 3 - 1;
        int dx  = tap % 3 - 1;
        const bf16_t* bpA = act + pidx(n, yy, xA + dx, H, W) * Cin + (khalf << 4);
        const bf16_t* bpB = act + pidx(n, yy, xB + dx, H, W) * Cin + (khalf << 4);
        __builtin_prefetch((const void*)bpA, 0, 1);   // global_prefetch_b8
        for (int j = 0; j < c16h; ++j, ++kb) {
            v16bf a  = *(const v16bf*)(wpo + (size_t)kb * 512);
            v16bf bA = *(const v16bf*)(bpA + (j << 5));
            v16bf bB = *(const v16bf*)(bpB + (j << 5));
            accA = __builtin_amdgcn_wmma_f32_16x16x32_bf16(
                false, a, false, bA, (short)0, accA, false, false);
            accB = __builtin_amdgcn_wmma_f32_16x16x32_bf16(
                false, a, false, bB, (short)0, accB, false, false);
        }
    }
#pragma unroll
    for (int r = 0; r < 8; ++r) {
        int m  = r + (khalf << 3);
        int oc = oct * 16 + m;
        if (oc < OC) {
            float bval = bias[oc];
            float vA = accA[r] + bval;
            float vB = accB[r] + bval;
            if (relu) { vA = fmaxf(vA, 0.f); vB = fmaxf(vB, 0.f); }
            size_t oiA = ((size_t)(n * OC + oc) * H + y) * W + xA;
            size_t oiB = oiA + 16;
            if (residual) { vA += residual[oiA]; vB += residual[oiB]; }
            if (outF32) { outF32[oiA] = vA; outF32[oiB] = vB; }
            if (outBf) {
                outBf[pidx(n, y, xA, H, W) * OC + oc] = f2bf(vA);
                outBf[pidx(n, y, xB, H, W) * OC + oc] = f2bf(vB);
            }
        }
    }
}

// ---------------------------------------------------------------------------
// Deformable 3x3 conv. One wave owns 16 pixels of one row:
//  Phase 1: branchless bilinear gather (corners clamped into the zero halo,
//           matching the reference's validity masking), sigmoid(mask),
//           staged as bf16 in LDS [pixel][tap*64+c] via b128 stores.
//  Phase 2: 64x576x16 GEMM: v_wmma, A from packed weights, B from LDS
//           (ds_load_b128).
// ---------------------------------------------------------------------------
__global__ __launch_bounds__(64)
void k_deform(const bf16_t* __restrict__ actb, int chanBase,
              const float* __restrict__ off,          // 27-ch NCHW (dy,dx x9, mask x9)
              const bf16_t* __restrict__ wp, const float* __restrict__ bias,
              float* __restrict__ outF32, int N, int H, int W) {
    __shared__ __attribute__((aligned(32))) bf16_t samp[2][16][576];
    int wid  = threadIdx.x >> 5;
    int lane = threadIdx.x & 31;
    int tile = blockIdx.x * 2 + wid;
    int HW   = H * W;
    int pix0 = tile << 4;
    int n    = pix0 / HW;
    int rem  = pix0 - n * HW;
    int y    = rem / W;
    int x0   = rem - y * W;

    // ---- phase 1: gather 16 pixels x 9 taps, 64 ch each ----
    for (int u = lane; u < 144; u += 32) {
        int p = u & 15, t = u >> 4;
        int x = x0 + p;
        size_t ob = ((size_t)(n * 27 + 2 * t) * H + y) * W + x;
        float dy  = off[ob];
        float dx  = off[ob + HW];
        float mm  = off[((size_t)(n * 27 + 18 + t) * H + y) * W + x];
        float msk = 1.f / (1.f + __expf(-mm));
        float py = dy + (float)(t / 3 - 1) + (float)y;
        float px = dx + (float)(t % 3 - 1) + (float)x;
        float fy = floorf(py), fx = floorf(px);
        float wy = py - fy, wx = px - fx;
        int iy0 = (int)fy, ix0 = (int)fx;
        float w00 = (1.f - wy) * (1.f - wx) * msk;
        float w01 = (1.f - wy) * wx * msk;
        float w10 = wy * (1.f - wx) * msk;
        float w11 = wy * wx * msk;
        // clamp each corner into the zero halo: OOB corners read 0 exactly
        // like the reference's validity mask.
        int y0c = min(max(iy0,     -1), H);
        int y1c = min(max(iy0 + 1, -1), H);
        int x0c = min(max(ix0,     -1), W);
        int x1c = min(max(ix0 + 1, -1), W);
        const bf16_t* base = actb + chanBase;
        const bf16_t* p00 = base + pidx(n, y0c, x0c, H, W) * 192;
        const bf16_t* p01 = base + pidx(n, y0c, x1c, H, W) * 192;
        const bf16_t* p10 = base + pidx(n, y1c, x0c, H, W) * 192;
        const bf16_t* p11 = base + pidx(n, y1c, x1c, H, W) * 192;
        bf16_t* dst = &samp[wid][p][t * 64];
#pragma unroll
        for (int cc = 0; cc < 4; ++cc) {
            v16bf q00 = *(const v16bf*)(p00 + cc * 16);
            v16bf q01 = *(const v16bf*)(p01 + cc * 16);
            v16bf q10 = *(const v16bf*)(p10 + cc * 16);
            v16bf q11 = *(const v16bf*)(p11 + cc * 16);
            v16bf r;
#pragma unroll
            for (int i = 0; i < 16; ++i) {
                float s = w00 * bf2f(q00[i]) + w01 * bf2f(q01[i])
                        + w10 * bf2f(q10[i]) + w11 * bf2f(q11[i]);
                r[i] = f2bf(s);
            }
            *(v16bf*)(dst + cc * 16) = r;      // ds_store_b128 x2
        }
    }
    __syncthreads();

    // ---- phase 2: GEMM 64oc x (9*64)K x 16px ----
    int col = lane & 15, khalf = lane >> 4;
    int x = x0 + col;
#pragma unroll
    for (int oct = 0; oct < 4; ++oct) {
        v8f acc = {};
        const bf16_t* wpo = wp + (size_t)oct * 18 * 512 + lane * 16;
        for (int kb = 0; kb < 18; ++kb) {
            v16bf a = *(const v16bf*)(wpo + (size_t)kb * 512);
            int k16 = kb * 2 + khalf;
            v16bf b = *(const v16bf*)(&samp[wid][col][k16 * 16]);   // ds_load_b128 x2
            acc = __builtin_amdgcn_wmma_f32_16x16x32_bf16(
                false, a, false, b, (short)0, acc, false, false);
        }
#pragma unroll
        for (int r = 0; r < 8; ++r) {
            int m  = r + (khalf << 3);
            int oc = oct * 16 + m;
            outF32[((size_t)(n * 64 + oc) * H + y) * W + x] = acc[r] + bias[oc];
        }
    }
}

// ---------------------------------------------------------------------------
// Host-side orchestration (graph-capture safe: only kernel launches on stream)
// ---------------------------------------------------------------------------
extern "C" void kernel_launch(void* const* d_in, const int* in_sizes, int n_in,
                              void* d_out, int out_size, void* d_ws, size_t ws_size,
                              hipStream_t stream) {
    (void)in_sizes; (void)n_in; (void)out_size; (void)ws_size;
    const int N = 2, H = 128, W = 128;
    const size_t HW = (size_t)H * W;
    const size_t PH = (size_t)(H + 2) * (W + 2);   // halo-padded pixels per image

    // inputs (setup_inputs dict order)
    const float* xd_in  = (const float*)d_in[0];
    const float* xl_in  = (const float*)d_in[1];
    const float* xr_in  = (const float*)d_in[2];
    const float* up_w   = (const float*)d_in[3];
    const float* up_b   = (const float*)d_in[4];
    const float* offl_w = (const float*)d_in[5];
    const float* offl_b = (const float*)d_in[6];
    const float* dl_w   = (const float*)d_in[7];
    const float* dl_b   = (const float*)d_in[8];
    const float* offr_w = (const float*)d_in[9];
    const float* offr_b = (const float*)d_in[10];
    const float* dr_w   = (const float*)d_in[11];
    const float* dr_b   = (const float*)d_in[12];
    const float* cv_w   = (const float*)d_in[13];
    const float* cv_b   = (const float*)d_in[14];
    const float* rb1_w1 = (const float*)d_in[15];
    const float* rb1_w2 = (const float*)d_in[16];
    const float* rb2_w1 = (const float*)d_in[17];
    const float* rb2_w2 = (const float*)d_in[18];
    const float* rb1_b1 = (const float*)d_in[19];
    const float* rb1_b2 = (const float*)d_in[20];
    const float* rb2_b1 = (const float*)d_in[21];
    const float* rb2_b2 = (const float*)d_in[22];

    // workspace layout (all 256B aligned; total ~61 MB)
    char* ws = (char*)d_ws;
    size_t off = 0;
    bf16_t* B0  = (bf16_t*)(ws + off); off += ((N * PH * 192 * 2 + 255) & ~255ull);
    float*  B2  = (float*) (ws + off); off += N * 27 * HW * 4;
    float*  B3  = (float*) (ws + off); off += N * 64 * HW * 4;
    float*  B4  = (float*) (ws + off); off += N * 64 * HW * 4;
    float*  B5  = (float*) (ws + off); off += N * 64 * HW * 4;
    float*  B5b = (float*) (ws + off); off += N * 64 * HW * 4;
    bf16_t* B6a = (bf16_t*)(ws + off); off += ((N * PH * 64 * 2 + 255) & ~255ull);
    bf16_t* B6b = (bf16_t*)(ws + off); off += ((N * PH * 64 * 2 + 255) & ~255ull);
    bf16_t* WPB = (bf16_t*)(ws + off);
    size_t wo = 0;
    bf16_t* wp_offl = WPB + wo; wo += (size_t)2 * 54 * 512;
    bf16_t* wp_offr = WPB + wo; wo += (size_t)2 * 54 * 512;
    bf16_t* wp_dl   = WPB + wo; wo += (size_t)4 * 18 * 512;
    bf16_t* wp_dr   = WPB + wo; wo += (size_t)4 * 18 * 512;
    bf16_t* wp_cv   = WPB + wo; wo += (size_t)4 * 54 * 512;
    bf16_t* wp_r1a  = WPB + wo; wo += (size_t)4 * 18 * 512;
    bf16_t* wp_r1b  = WPB + wo; wo += (size_t)4 * 18 * 512;
    bf16_t* wp_r2a  = WPB + wo; wo += (size_t)4 * 18 * 512;
    bf16_t* wp_r2b  = WPB + wo; wo += (size_t)4 * 18 * 512;

    auto packw = [&](const float* w, bf16_t* wp, int O, int Cin) {
        int total = ((O + 15) / 16) * ((Cin * 9) / 32) * 512;
        k_packw<<<(total + 255) / 256, 256, 0, stream>>>(w, wp, O, Cin);
    };
    auto conv = [&](const bf16_t* act, int Cin, const bf16_t* wp, const float* bias,
                    int OC, int relu, const float* res, float* of, bf16_t* ob) {
        int OCT = (OC + 15) / 16;
        int blocks = (N * H * (W / 32) * OCT + 7) / 8;
        k_conv3x3<<<blocks, 256, 0, stream>>>(act, Cin, wp, bias, OC, OCT, relu,
                                              res, of, ob, N, H, W);
    };

    // 0) pre-swizzle all conv weights; zero activation halos (deterministic)
    packw(offl_w, wp_offl, 27, 192);
    packw(offr_w, wp_offr, 27, 192);
    packw(dl_w,   wp_dl,   64, 64);
    packw(dr_w,   wp_dr,   64, 64);
    packw(cv_w,   wp_cv,   64, 192);
    packw(rb1_w1, wp_r1a,  64, 64);
    packw(rb1_w2, wp_r1b,  64, 64);
    packw(rb2_w1, wp_r2a,  64, 64);
    packw(rb2_w2, wp_r2b,  64, 64);
    int hb = (int)((N * PH + 255) / 256);
    k_zero_halo<<<hb, 256, 0, stream>>>(B0,  192, N, H, W);
    k_zero_halo<<<hb, 256, 0, stream>>>(B6a, 64,  N, H, W);
    k_zero_halo<<<hb, 256, 0, stream>>>(B6b, 64,  N, H, W);

    // 1) build NHWC(halo) bf16 concat [xl | xr | upsampled-conv(xd)]
    k_pack3<<<(int)((N * HW * 128 + 255) / 256), 256, 0, stream>>>(xl_in, xr_in, B0, N, H, W);
    k_upconv<<<(int)(N * HW / 256) * 4, 256, 0, stream>>>(xd_in, up_w, up_b, B0, N, H, W);

    // 2) left offsets + deformable conv, refresh concat slice
    conv(B0, 192, wp_offl, offl_b, 27, 0, nullptr, B2, nullptr);
    k_deform<<<(int)(N * HW / 16 / 2), 64, 0, stream>>>(B0, 0, B2, wp_dl, dl_b, B3, N, H, W);
    k_pack1<<<(int)((N * HW * 64 + 255) / 256), 256, 0, stream>>>(B3, B0, 0, N, H, W);

    // 3) right offsets + deformable conv, refresh concat slice
    conv(B0, 192, wp_offr, offr_b, 27, 0, nullptr, B2, nullptr);
    k_deform<<<(int)(N * HW / 16 / 2), 64, 0, stream>>>(B0, 64, B2, wp_dr, dr_b, B4, N, H, W);
    k_pack1<<<(int)((N * HW * 64 + 255) / 256), 256, 0, stream>>>(B4, B0, 64, N, H, W);

    // 4) fuse conv + ReLU; keep f32 (residual) and bf16 NHWC(halo) chain input
    conv(B0, 192, wp_cv, cv_b, 64, 1, nullptr, B5, B6a);

    // 5) residual block 1: t=relu(conv(x)); x = x + conv(t)
    conv(B6a, 64, wp_r1a, rb1_b1, 64, 1, nullptr, nullptr, B6b);
    conv(B6b, 64, wp_r1b, rb1_b2, 64, 0, B5, B5b, B6a);

    // 6) residual block 2 -> final f32 NCHW output
    conv(B6a, 64, wp_r2a, rb2_b1, 64, 1, nullptr, nullptr, B6b);
    conv(B6b, 64, wp_r2b, rb2_b2, 64, 0, B5b, (float*)d_out, nullptr);
}
